// Model_36910948942195
// MI455X (gfx1250) — matrix operand order repaired and checked by reading, not compiled, not run on hardware
//
#include <hip/hip_runtime.h>

// ---------------------------------------------------------------------------
// Model dims (fixed by the reference)
// ---------------------------------------------------------------------------
static constexpr int Bb = 32, Tt = 4096, Dm = 128, Hh = 4, dh = 32, Ll = 4, Vv = 33;
static constexpr int ROWS = Bb * Tt; // 131072

typedef __attribute__((ext_vector_type(16))) __bf16          v16bf;
typedef __attribute__((ext_vector_type(8)))  float           v8f;
typedef __attribute__((ext_vector_type(16))) unsigned short  v16u;
typedef __attribute__((ext_vector_type(8)))  unsigned short  v8u;
typedef __attribute__((ext_vector_type(4)))  unsigned int    v4ui;
typedef __attribute__((ext_vector_type(8)))  int             v8i;
typedef __attribute__((ext_vector_type(4)))  int             v4i;

union FragBF { v16u u; v16bf b; };

// ---------------------------------------------------------------------------
// Helpers
// ---------------------------------------------------------------------------
__device__ __forceinline__ unsigned short f2b(float f) {
  union { float f; unsigned u; } x; x.f = f;
  unsigned r = x.u + 0x7FFFu + ((x.u >> 16) & 1u); // round-to-nearest-even
  return (unsigned short)(r >> 16);
}
__device__ __forceinline__ float b2f(unsigned short h) {
  union { unsigned u; float f; } x; x.u = ((unsigned)h) << 16;
  return x.f;
}
__device__ __forceinline__ float silu_f(float x) { return x / (1.f + __expf(-x)); }
__device__ __forceinline__ float sigm_f(float x) { return 1.f / (1.f + __expf(-x)); }
__device__ __forceinline__ int   lane_id() { return (int)__builtin_amdgcn_mbcnt_lo(~0u, 0u); }

__device__ __forceinline__ float bcast32(float v, int src) {
  int i = __builtin_amdgcn_ds_bpermute(src << 2, __builtin_bit_cast(int, v));
  return __builtin_bit_cast(float, i);
}
__device__ __forceinline__ float red_sum32(float v) {
  int l = lane_id();
#pragma unroll
  for (int m = 16; m >= 1; m >>= 1) {
    int o = __builtin_amdgcn_ds_bpermute(((l ^ m) << 2), __builtin_bit_cast(int, v));
    v += __builtin_bit_cast(float, o);
  }
  return v;
}

// ---------------------------------------------------------------------------
// f32 -> bf16 convert (weights, activations)
// ---------------------------------------------------------------------------
__global__ void k_f32_to_bf16(const float* __restrict__ in, unsigned short* __restrict__ out, int n) {
  int i = blockIdx.x * 256 + threadIdx.x;
  if (i < n) out[i] = f2b(in[i]);
}

// ---------------------------------------------------------------------------
// embedding gather: h[row, c] = emb[x[row], c]
// ---------------------------------------------------------------------------
__global__ void k_embed(const int* __restrict__ x, const float* __restrict__ emb,
                        float* __restrict__ h, int n) {
  int i = blockIdx.x * 256 + threadIdx.x;
  if (i < n) {
    int row = i >> 7, c = i & 127;
    h[i] = emb[x[row] * Dm + c];
  }
}

// ---------------------------------------------------------------------------
// RMSNorm over D=128, output bf16 (one wave per row, float4 per lane)
// ---------------------------------------------------------------------------
__global__ void k_rms_bf16(const float* __restrict__ in, const float* __restrict__ w,
                           unsigned short* __restrict__ out, int rows) {
  int wave = threadIdx.x >> 5, lane = threadIdx.x & 31;
  int row = blockIdx.x * 8 + wave;
  if (row >= rows) return;
  const float4 xv = *(const float4*)(in + (size_t)row * Dm + lane * 4);
  float s = xv.x * xv.x + xv.y * xv.y + xv.z * xv.z + xv.w * xv.w;
  s = red_sum32(s);
  float sc = rsqrtf(s * (1.f / 128.f) + 1e-5f);
  const float4 wv = *(const float4*)(w + lane * 4);
  ushort4 o;
  o.x = f2b(xv.x * sc * wv.x);
  o.y = f2b(xv.y * sc * wv.y);
  o.z = f2b(xv.z * sc * wv.z);
  o.w = f2b(xv.w * sc * wv.w);
  *(ushort4*)(out + (size_t)row * Dm + lane * 4) = o;
}

// ---------------------------------------------------------------------------
// WMMA GEMM: C[M,N] (f32) = A[M,K](bf16) * W[N,K](bf16)^T  (+ C if accumulate)
// block = 256 threads (8 waves). Block tile: 128(M) x 128(N).
// W tile DMA'd into LDS by the Tensor Data Mover (wave 0 issues a 2D D#;
// TDM OOB reads return zero, which zero-pads the V=33 head tile for free).
// Each wave: 16 M-rows x eight 16-col tiles; A-frags software-pipelined.
// ---------------------------------------------------------------------------
__global__ void k_gemm_bf16(const unsigned short* __restrict__ A,
                            const unsigned short* __restrict__ W,
                            float* __restrict__ C,
                            int M, int N, int K, int accumulate) {
  extern __shared__ unsigned short Wlds[];
  const int BN = 128;
  const int tid = threadIdx.x;
  const int wave = tid >> 5, lane = tid & 31;
  const int m0 = blockIdx.x * 128 + wave * 16;
  const int n0 = blockIdx.y * BN;

  // ---- TDM: stage W tile [BN x K] (bf16) global -> LDS --------------------
  if (wave == 0) {
    unsigned long long ga = (unsigned long long)(size_t)(W + (size_t)n0 * K);
    unsigned lds_off = (unsigned)(size_t)(void*)Wlds;
    unsigned td1 = (unsigned)(N - n0);      // remaining rows; OOB rows read 0
    v4ui g0;
    g0[0] = 1u;                              // count=1, user descriptor
    g0[1] = lds_off;                         // lds_addr
    g0[2] = (unsigned)(ga & 0xFFFFFFFFu);    // global_addr[31:0]
    g0[3] = (unsigned)((ga >> 32) & 0x01FFFFFFu) | 0x80000000u; // addr[56:32] | type=2
    v8i g1;
    g1[0] = (int)(1u << 16);                 // data_size=1 (2 bytes)
    g1[1] = (int)(((unsigned)K & 0xFFFFu) << 16);                 // tensor_dim0 lo
    g1[2] = (int)((((unsigned)K >> 16) & 0xFFFFu) | ((td1 & 0xFFFFu) << 16)); // dim0 hi | dim1 lo
    g1[3] = (int)(((td1 >> 16) & 0xFFFFu) | (((unsigned)K & 0xFFFFu) << 16)); // dim1 hi | tile_dim0=K
    g1[4] = (int)((unsigned)BN & 0xFFFFu);   // tile_dim1=BN, tile_dim2=0
    g1[5] = (int)(unsigned)K;                // tensor_dim0_stride[31:0] = K
    g1[6] = 0;                               // stride hi | dim1_stride lo
    g1[7] = 0;
    v4i gz4 = {0, 0, 0, 0};
    v8i gz8 = {0, 0, 0, 0, 0, 0, 0, 0};
    __builtin_amdgcn_tensor_load_to_lds(g0, g1, gz4, gz4, gz8, 0);
    __builtin_amdgcn_s_wait_tensorcnt(0);
  }
  __syncthreads();

  const int lr = lane & 15, hb = lane >> 4;
  v8f acc[8];
#pragma unroll
  for (int j = 0; j < 8; ++j) {
    int n = n0 + j * 16 + lr;
    if (accumulate && n < N) {
#pragma unroll
      for (int r = 0; r < 8; ++r)
        acc[j][r] = C[(size_t)(m0 + hb * 8 + r) * N + n];
    } else {
#pragma unroll
      for (int r = 0; r < 8; ++r) acc[j][r] = 0.f;
    }
  }

  // A-frag: lane<16 -> K k0+0..15 ; lane>=16 -> K k0+16..31 (matches B)
  const unsigned short* arow = A + (size_t)(m0 + lr) * K + hb * 16;
  auto loadA = [&](int k0) -> FragBF {
    v8u alo = *(const v8u*)(arow + k0);
    v8u ahi = *(const v8u*)(arow + k0 + 8);
    FragBF f;
#pragma unroll
    for (int i = 0; i < 8; ++i) { f.u[i] = alo[i]; f.u[8 + i] = ahi[i]; }
    return f;
  };

  FragBF fa = loadA(0);
  for (int k0 = 0; k0 < K; k0 += 32) {
    FragBF fn = fa;
    if (k0 + 32 < K) {
      fn = loadA(k0 + 32);                       // pipeline: overlap with wmma chain
      __builtin_prefetch((const void*)(arow + k0 + 64), 0, 3);
    }
#pragma unroll
    for (int j = 0; j < 8; ++j) {
      // B-frag: lane<16 -> col n, K k0+0..15 ; lane>=16 -> col n, K k0+16..31
      FragBF fb;
      fb.u = *(const v16u*)(&Wlds[(size_t)(j * 16 + lr) * K + k0 + hb * 16]);
      acc[j] = __builtin_amdgcn_wmma_f32_16x16x32_bf16(
          false, fa.b, false, fb.b, (short)0, acc[j], false, false);
    }
    fa = fn;
  }

#pragma unroll
  for (int j = 0; j < 8; ++j) {
    int n = n0 + j * 16 + lr;
    if (n < N) {
#pragma unroll
      for (int r = 0; r < 8; ++r)
        C[(size_t)(m0 + hb * 8 + r) * N + n] = acc[j][r];
    }
  }
}

// ---------------------------------------------------------------------------
// Depthwise causal conv (K=4) + SiLU, then optional per-head L2-norm.
// mode: 0 = q (l2n * 1/sqrt(32)), 1 = k (l2n), 2 = v (none).
// block = 128 threads (thread == channel); wave == head (32 contiguous chans).
// ---------------------------------------------------------------------------
__global__ void k_conv(const float* __restrict__ in, const float* __restrict__ cw,
                       const float* __restrict__ cb, float* __restrict__ out, int mode) {
  const int c = threadIdx.x;
  const int b = blockIdx.x;
  const int t0 = blockIdx.y * 512;
  const float w0 = cw[c * 4 + 0], w1 = cw[c * 4 + 1];
  const float w2 = cw[c * 4 + 2], w3 = cw[c * 4 + 3];
  const float bias = cb[c];
  const size_t base = ((size_t)b * Tt) * Dm + c;
  float xm1 = (t0 >= 1) ? in[base + (size_t)(t0 - 1) * Dm] : 0.f;
  float xm2 = (t0 >= 2) ? in[base + (size_t)(t0 - 2) * Dm] : 0.f;
  float xm3 = (t0 >= 3) ? in[base + (size_t)(t0 - 3) * Dm] : 0.f;
  const float qs = 0.17677669529663687f; // 1/sqrt(32)
  for (int t = t0; t < t0 + 512; ++t) {
    float x0 = in[base + (size_t)t * Dm];
    float y = w0 * xm3 + w1 * xm2 + w2 * xm1 + w3 * x0 + bias;
    y = silu_f(y);
    if (mode < 2) {
      float ss = red_sum32(y * y);
      y *= rsqrtf(ss + 1e-6f);
      if (mode == 0) y *= qs;
    }
    out[base + (size_t)t * Dm] = y;
    xm3 = xm2; xm2 = xm1; xm1 = x0;
  }
}

// ---------------------------------------------------------------------------
// beta[row, h] = sigmoid(dot(hn_bf16[row,:], bw[h,:]))   (N=4 -> scalar kernel)
// ---------------------------------------------------------------------------
__global__ void k_beta(const unsigned short* __restrict__ hn, const float* __restrict__ bw,
                       float* __restrict__ beta, int total) {
  int i = blockIdx.x * 256 + threadIdx.x;
  if (i >= total) return;
  int row = i >> 2, hh = i & 3;
  const v8u* hp = (const v8u*)(hn + (size_t)row * Dm);
  const float* wp = bw + hh * Dm;
  float s = 0.f;
#pragma unroll
  for (int cc = 0; cc < 16; ++cc) {
    v8u u = hp[cc];
#pragma unroll
    for (int j = 0; j < 8; ++j) s += b2f(u[j]) * wp[cc * 8 + j];
  }
  beta[i] = sigm_f(s);
}

// ---------------------------------------------------------------------------
// Rank-1 delta-rule scan. One wave32 per (b,h). State S[32][32]:
// lane j holds column j (S[i][j] for i=0..31) in registers.
// ---------------------------------------------------------------------------
__global__ void k_delta_scan(const float* __restrict__ q, const float* __restrict__ k,
                             const float* __restrict__ v, const float* __restrict__ beta,
                             float* __restrict__ o) {
  const int bh = blockIdx.x;
  const int b = bh >> 2, hh = bh & 3;
  const int j = threadIdx.x; // 0..31
  float S[32];
#pragma unroll
  for (int i = 0; i < 32; ++i) S[i] = 0.f;
  const size_t cbase = (size_t)b * Tt * Dm + hh * 32 + j;
  const size_t bbase = (size_t)b * Tt * Hh + hh;
  for (int t = 0; t < Tt; ++t) {
    size_t idx = cbase + (size_t)t * Dm;
    float qv = q[idx], kv = k[idx], vv = v[idx];
    float bt = beta[bbase + (size_t)t * Hh];
    float dot = 0.f;
#pragma unroll
    for (int i = 0; i < 32; ++i) dot += S[i] * bcast32(kv, i);
    float err = (vv - dot) * bt;
    float outv = 0.f;
#pragma unroll
    for (int i = 0; i < 32; ++i) S[i] += bcast32(kv, i) * err;
#pragma unroll
    for (int i = 0; i < 32; ++i) outv += S[i] * bcast32(qv, i);
    o[idx] = outv;
  }
}

// ---------------------------------------------------------------------------
// o_bf16 = rms(o over d=32, hnw) * silu(gate).  One wave per (row, head).
// ---------------------------------------------------------------------------
__global__ void k_gate_out(const float* __restrict__ o, const float* __restrict__ gate,
                           const float* __restrict__ hnw, unsigned short* __restrict__ ob,
                           int totalWaves) {
  int wave = threadIdx.x >> 5, lane = threadIdx.x & 31;
  int g = blockIdx.x * 8 + wave;
  if (g >= totalWaves) return;
  int row = g >> 2, hh = g & 3;
  size_t idx = (size_t)row * Dm + hh * 32 + lane;
  float ov = o[idx];
  float ss = red_sum32(ov * ov);
  float sc = rsqrtf(ss * (1.f / 32.f) + 1e-5f);
  ob[idx] = f2b(ov * sc * hnw[lane] * silu_f(gate[idx]));
}

// ---------------------------------------------------------------------------
// elementwise silu -> bf16 (MLP mid)
// ---------------------------------------------------------------------------
__global__ void k_silu_bf16(const float* __restrict__ in, unsigned short* __restrict__ out, int n) {
  int i = blockIdx.x * 256 + threadIdx.x;
  if (i < n) out[i] = f2b(silu_f(in[i]));
}

// ---------------------------------------------------------------------------
// Host orchestration
// ---------------------------------------------------------------------------
extern "C" void kernel_launch(void* const* d_in, const int* in_sizes, int n_in,
                              void* d_out, int out_size, void* d_ws, size_t ws_size,
                              hipStream_t stream) {
  (void)in_sizes; (void)n_in; (void)out_size; (void)ws_size;
  const int*   x   = (const int*)d_in[0];
  const float* emb = (const float*)d_in[1];
  const float* qw  = (const float*)d_in[2];
  const float* kw  = (const float*)d_in[3];
  const float* vw  = (const float*)d_in[4];
  const float* qcw = (const float*)d_in[5];
  const float* qcb = (const float*)d_in[6];
  const float* kcw = (const float*)d_in[7];
  const float* kcb = (const float*)d_in[8];
  const float* vcw = (const float*)d_in[9];
  const float* vcb = (const float*)d_in[10];
  const float* ow  = (const float*)d_in[11];
  const float* gw  = (const float*)d_in[12];
  const float* bw  = (const float*)d_in[13];
  const float* hnw = (const float*)d_in[14];
  const float* n1w = (const float*)d_in[15];
  const float* n2w = (const float*)d_in[16];
  const float* m1w = (const float*)d_in[17];
  const float* m2w = (const float*)d_in[18];
  const float* fnw = (const float*)d_in[19];
  const float* hw  = (const float*)d_in[20];
  float* outp = (float*)d_out;

  const int rows = ROWS;
  char* p = (char*)d_ws;
  auto take = [&](size_t bytes) -> char* {
    char* r = p; p += (bytes + 255) & ~(size_t)255; return r;
  };
  float* h    = (float*)take((size_t)rows * Dm * 4);
  float* lin  = (float*)take((size_t)rows * Dm * 4); // gemm out scratch; holds gate
  float* qb   = (float*)take((size_t)rows * Dm * 4);
  float* kb   = (float*)take((size_t)rows * Dm * 4);
  float* vb   = (float*)take((size_t)rows * Dm * 4);
  float* obuf = (float*)take((size_t)rows * Dm * 4);
  float* mid  = (float*)take((size_t)rows * 2 * Dm * 4);
  unsigned short* bfa  = (unsigned short*)take((size_t)rows * 2 * Dm * 2); // o_bf16 / mid_bf16
  unsigned short* bfh  = (unsigned short*)take((size_t)rows * Dm * 2);     // hn / hn2 / final
  float* beta = (float*)take((size_t)rows * Hh * 4);
  unsigned short* wqb  = (unsigned short*)take((size_t)Ll * Dm * Dm * 2);
  unsigned short* wkb  = (unsigned short*)take((size_t)Ll * Dm * Dm * 2);
  unsigned short* wvb  = (unsigned short*)take((size_t)Ll * Dm * Dm * 2);
  unsigned short* wgb  = (unsigned short*)take((size_t)Ll * Dm * Dm * 2);
  unsigned short* wob  = (unsigned short*)take((size_t)Ll * Dm * Dm * 2);
  unsigned short* wm1b = (unsigned short*)take((size_t)Ll * 2 * Dm * Dm * 2);
  unsigned short* wm2b = (unsigned short*)take((size_t)Ll * 2 * Dm * Dm * 2);
  unsigned short* whwb = (unsigned short*)take((size_t)Vv * Dm * 2);

  auto cvt = [&](const float* src, unsigned short* dst, int n) {
    k_f32_to_bf16<<<(n + 255) / 256, 256, 0, stream>>>(src, dst, n);
  };
  cvt(qw, wqb, Ll * Dm * Dm);
  cvt(kw, wkb, Ll * Dm * Dm);
  cvt(vw, wvb, Ll * Dm * Dm);
  cvt(gw, wgb, Ll * Dm * Dm);
  cvt(ow, wob, Ll * Dm * Dm);
  cvt(m1w, wm1b, Ll * 2 * Dm * Dm);
  cvt(m2w, wm2b, Ll * 2 * Dm * Dm);
  cvt(hw, whwb, Vv * Dm);

  auto gemm = [&](const unsigned short* A, const unsigned short* Wt, float* C,
                  int M, int N, int K, int acc) {
    dim3 grid(M / 128, (N + 127) / 128);
    k_gemm_bf16<<<grid, 256, (size_t)128 * K * 2, stream>>>(A, Wt, C, M, N, K, acc);
  };

  k_embed<<<(rows * Dm) / 256, 256, 0, stream>>>(x, emb, h, rows * Dm);

  for (int l = 0; l < Ll; ++l) {
    const int wOff = l * Dm * Dm, mOff = l * 2 * Dm * Dm;
    k_rms_bf16<<<rows / 8, 256, 0, stream>>>(h, n1w + l * Dm, bfh, rows);

    gemm(bfh, wqb + wOff, lin, rows, Dm, Dm, 0);
    k_conv<<<dim3(Bb, Tt / 512), 128, 0, stream>>>(lin, qcw + l * Dm * 4, qcb + l * Dm, qb, 0);
    gemm(bfh, wkb + wOff, lin, rows, Dm, Dm, 0);
    k_conv<<<dim3(Bb, Tt / 512), 128, 0, stream>>>(lin, kcw + l * Dm * 4, kcb + l * Dm, kb, 1);
    gemm(bfh, wvb + wOff, lin, rows, Dm, Dm, 0);
    k_conv<<<dim3(Bb, Tt / 512), 128, 0, stream>>>(lin, vcw + l * Dm * 4, vcb + l * Dm, vb, 2);
    gemm(bfh, wgb + wOff, lin, rows, Dm, Dm, 0); // gate lives in lin
    k_beta<<<(rows * Hh) / 256, 256, 0, stream>>>(bfh, bw + l * Hh * Dm, beta, rows * Hh);

    k_delta_scan<<<Bb * Hh, 32, 0, stream>>>(qb, kb, vb, beta, obuf);
    k_gate_out<<<(rows * Hh) / 8, 256, 0, stream>>>(obuf, lin, hnw + l * dh, bfa, rows * Hh);
    gemm(bfa, wob + wOff, h, rows, Dm, Dm, 1); // h += o @ ow^T

    k_rms_bf16<<<rows / 8, 256, 0, stream>>>(h, n2w + l * Dm, bfh, rows);
    gemm(bfh, wm1b + mOff, mid, rows, 2 * Dm, Dm, 0);
    k_silu_bf16<<<(rows * 2 * Dm) / 256, 256, 0, stream>>>(mid, bfa, rows * 2 * Dm);
    gemm(bfa, wm2b + mOff, h, rows, Dm, 2 * Dm, 1); // h += mid @ m2^T
  }

  k_rms_bf16<<<rows / 8, 256, 0, stream>>>(h, fnw, bfh, rows);
  gemm(bfh, whwb, outp, rows, Vv, Dm, 0);
}
